// SphericalHarmonicTransform_335007449425
// MI455X (gfx1250) — matrix operand order repaired and checked by reading, not compiled, not run on hardware
//
#include <hip/hip_runtime.h>
#include <hip/hip_bf16.h>

typedef __attribute__((ext_vector_type(16))) __bf16 v16bf;
typedef __attribute__((ext_vector_type(8)))  __bf16 v8bf;
typedef __attribute__((ext_vector_type(8)))  float  v8f;

#define NLAT   256
#define NLON   512
#define LMAX   256
#define MMAX   256
#define NBATCH 256   // 4 * 64
#define SHT_PI 3.14159265358979323846

__device__ __forceinline__ void split_bf16(float x, __bf16& h, __bf16& l) {
    h = (__bf16)x;
    l = (__bf16)(x - (float)h);
}

// ---------------------------------------------------------------------------
// Twiddle tables: ct[m][n] = (2pi/512) cos(-2pi m n / 512), st = (2pi/512) sin(-...)
// (imag part of e^{-i theta} directly, so no sign fix-ups later)
// ---------------------------------------------------------------------------
__global__ void sht_twiddle(__bf16* __restrict__ cth, __bf16* __restrict__ ctl,
                            __bf16* __restrict__ sth, __bf16* __restrict__ stl) {
    int idx = blockIdx.x * blockDim.x + threadIdx.x;
    if (idx >= MMAX * NLON) return;
    int m = idx >> 9;
    int n = idx & (NLON - 1);
    int p = (m * n) & (NLON - 1);                 // exact phase reduction
    double a = -2.0 * SHT_PI * (double)p / (double)NLON;
    double sc = 2.0 * SHT_PI / (double)NLON;
    float c = (float)(cos(a) * sc);
    float s = (float)(sin(a) * sc);
    split_bf16(c, cth[idx], ctl[idx]);
    split_bf16(s, sth[idx], stl[idx]);
}

// ---------------------------------------------------------------------------
// Weight split: fp32 [m][l][k] -> bf16 hi/lo, same layout (k contiguous = B operand)
// ---------------------------------------------------------------------------
__global__ void sht_wprep(const float* __restrict__ w,
                          __bf16* __restrict__ wh, __bf16* __restrict__ wl, int n) {
    int i = blockIdx.x * blockDim.x + threadIdx.x;
    if (i >= n) return;
    split_bf16(w[i], wh[i], wl[i]);
}

__device__ __forceinline__ v16bf cat8(v8bf a, v8bf b) {
    v16bf r;
#pragma unroll
    for (int e = 0; e < 8; ++e) { r[e] = a[e]; r[8 + e] = b[e]; }
    return r;
}

#define WMMA_BF16(A, B, C) \
    __builtin_amdgcn_wmma_f32_16x16x32_bf16(false, (A), false, (B), (short)0, (C), false, false)

// ---------------------------------------------------------------------------
// Stage 1: rFFT as GEMM.  Rows R = b*256 + k_lat (65536), cols m (256), red n (512).
// A = x (fp32, split hi/lo in-register), B = twiddles.  F written [m][b][k] bf16 hi/lo.
// Block = 256 thr = 8 waves; wave -> one 16x16 (R,m) tile.
// ---------------------------------------------------------------------------
__global__ __launch_bounds__(256)
void sht_dft(const float* __restrict__ x,
             const __bf16* __restrict__ cth, const __bf16* __restrict__ ctl,
             const __bf16* __restrict__ sth, const __bf16* __restrict__ stl,
             __bf16* __restrict__ frh, __bf16* __restrict__ frl,
             __bf16* __restrict__ fih, __bf16* __restrict__ fil) {
    const int lane = threadIdx.x & 31;
    const int wave = threadIdx.x >> 5;
    const int half = lane >> 4;
    const int l15  = lane & 15;
    const int R0 = blockIdx.x * 16;                  // 4096 row tiles
    const int m0 = (blockIdx.y * 8 + wave) * 16;     // 16 m tiles

    const float*  xrow = x   + (size_t)(R0 + l15) * NLON;       // A row (M = l15)
    const __bf16* bch  = cth + (size_t)(m0 + l15) * NLON + half * 16;  // B col (N = l15)
    const __bf16* bcl  = ctl + (size_t)(m0 + l15) * NLON + half * 16;
    const __bf16* bsh  = sth + (size_t)(m0 + l15) * NLON + half * 16;
    const __bf16* bsl  = stl + (size_t)(m0 + l15) * NLON + half * 16;

    v8f accR = {}; v8f accI = {};
    for (int n0 = 0; n0 < NLON; n0 += 32) {
        // A: K runs {n0+16h .. +15}, split fp32 -> bf16 hi/lo
        v8f a0 = *(const v8f*)(xrow + n0 + half * 16);
        v8f a1 = *(const v8f*)(xrow + n0 + half * 16 + 8);
        v16bf ah, al;
#pragma unroll
        for (int e = 0; e < 8; ++e) {
            float v = a0[e];
            __bf16 h = (__bf16)v;
            ah[e] = h;
            al[e] = (__bf16)(v - (float)h);
        }
#pragma unroll
        for (int e = 0; e < 8; ++e) {
            float v = a1[e];
            __bf16 h = (__bf16)v;
            ah[8 + e] = h;
            al[8 + e] = (__bf16)(v - (float)h);
        }

        v16bf Bch = *(const v16bf*)(bch + n0);
        v16bf Bcl = *(const v16bf*)(bcl + n0);
        v16bf Bsh = *(const v16bf*)(bsh + n0);
        v16bf Bsl = *(const v16bf*)(bsl + n0);

        accR = WMMA_BF16(ah, Bch, accR);
        accR = WMMA_BF16(ah, Bcl, accR);
        accR = WMMA_BF16(al, Bch, accR);
        accI = WMMA_BF16(ah, Bsh, accI);
        accI = WMMA_BF16(ah, Bsl, accI);
        accI = WMMA_BF16(al, Bsh, accI);
    }

#pragma unroll
    for (int r = 0; r < 8; ++r) {                    // D: row r+8h, col l15
        int R = R0 + r + half * 8;
        int b = R >> 8, k = R & 255;
        int m = m0 + l15;
        size_t o = ((size_t)m * NBATCH + b) * NLAT + k;
        split_bf16(accR[r], frh[o], frl[o]);
        split_bf16(accI[r], fih[o], fil[o]);
    }
}

// ---------------------------------------------------------------------------
// Stage 2: per-m Legendre GEMM.  C[b][l] = sum_k F[b][k] * W[m][l][k], re & im.
// B operand (weights) shared between the two accumulators.
// Grid (m=256, tilegroup=32), block 256 = 8 waves, wave -> one 16x16 (b,l) tile.
// ---------------------------------------------------------------------------
__global__ __launch_bounds__(256)
void sht_leg(const __bf16* __restrict__ frh, const __bf16* __restrict__ frl,
             const __bf16* __restrict__ fih, const __bf16* __restrict__ fil,
             const __bf16* __restrict__ wh,  const __bf16* __restrict__ wl,
             float* __restrict__ out) {
    const int lane = threadIdx.x & 31;
    const int wave = threadIdx.x >> 5;
    const int half = lane >> 4;
    const int l15  = lane & 15;
    const int m    = blockIdx.x;
    const int tile = blockIdx.y * 8 + wave;          // 0..255
    const int b0 = (tile >> 4) * 16;
    const int l0 = (tile & 15) * 16;

    const __bf16* arh = frh + ((size_t)m * NBATCH + b0 + l15) * NLAT;
    const __bf16* arl = frl + ((size_t)m * NBATCH + b0 + l15) * NLAT;
    const __bf16* aih = fih + ((size_t)m * NBATCH + b0 + l15) * NLAT;
    const __bf16* ail = fil + ((size_t)m * NBATCH + b0 + l15) * NLAT;
    const __bf16* bwh = wh  + ((size_t)m * LMAX  + l0 + l15) * NLAT + half * 16;
    const __bf16* bwl = wl  + ((size_t)m * LMAX  + l0 + l15) * NLAT + half * 16;

    v8f accR = {}; v8f accI = {};
    for (int k0 = 0; k0 < NLAT; k0 += 32) {
        const int ka = k0 + half * 16;
        __builtin_prefetch(bwh + k0 + 32, 0, 1);     // -> global_prefetch_b8
        v16bf Arh = cat8(*(const v8bf*)(arh + ka), *(const v8bf*)(arh + ka + 8));
        v16bf Arl = cat8(*(const v8bf*)(arl + ka), *(const v8bf*)(arl + ka + 8));
        v16bf Aih = cat8(*(const v8bf*)(aih + ka), *(const v8bf*)(aih + ka + 8));
        v16bf Ail = cat8(*(const v8bf*)(ail + ka), *(const v8bf*)(ail + ka + 8));
        v16bf Bh  = *(const v16bf*)(bwh + k0);
        v16bf Bl  = *(const v16bf*)(bwl + k0);

        accR = WMMA_BF16(Arh, Bh, accR);
        accR = WMMA_BF16(Arh, Bl, accR);
        accR = WMMA_BF16(Arl, Bh, accR);
        accI = WMMA_BF16(Aih, Bh, accI);
        accI = WMMA_BF16(Aih, Bl, accI);
        accI = WMMA_BF16(Ail, Bh, accI);
    }

#pragma unroll
    for (int r = 0; r < 8; ++r) {
        int b = b0 + r + half * 8;
        int l = l0 + l15;
        size_t o = (((size_t)b * LMAX + l) * MMAX + m) * 2;   // interleaved complex
        float2 v; v.x = accR[r]; v.y = accI[r];
        *(float2*)(out + o) = v;
    }
}

// ---------------------------------------------------------------------------
extern "C" void kernel_launch(void* const* d_in, const int* in_sizes, int n_in,
                              void* d_out, int out_size, void* d_ws, size_t ws_size,
                              hipStream_t stream) {
    const float* x = (const float*)d_in[0];   // (4,64,NLAT*NLON) fp32
    const float* w = (const float*)d_in[1];   // (MMAX,LMAX,NLAT) fp32
    float* out = (float*)d_out;               // (4,64,LMAX,MMAX) complex64 interleaved

    char* ws = (char*)d_ws;
    size_t off = 0;
    const size_t SZ_TW = (size_t)MMAX * NLON * sizeof(__bf16);          // 256 KB
    const size_t SZ_W  = (size_t)MMAX * LMAX * NLAT * sizeof(__bf16);   // 32 MB
    const size_t SZ_F  = (size_t)MMAX * NBATCH * NLAT * sizeof(__bf16); // 32 MB
    __bf16* cth = (__bf16*)(ws + off); off += SZ_TW;
    __bf16* ctl = (__bf16*)(ws + off); off += SZ_TW;
    __bf16* sth = (__bf16*)(ws + off); off += SZ_TW;
    __bf16* stl = (__bf16*)(ws + off); off += SZ_TW;
    __bf16* wh  = (__bf16*)(ws + off); off += SZ_W;
    __bf16* wl  = (__bf16*)(ws + off); off += SZ_W;
    __bf16* frh = (__bf16*)(ws + off); off += SZ_F;
    __bf16* frl = (__bf16*)(ws + off); off += SZ_F;
    __bf16* fih = (__bf16*)(ws + off); off += SZ_F;
    __bf16* fil = (__bf16*)(ws + off); off += SZ_F;
    // total ws use ~193 MB

    sht_twiddle<<<(MMAX * NLON + 255) / 256, 256, 0, stream>>>(cth, ctl, sth, stl);
    sht_wprep<<<(MMAX * LMAX * NLAT + 255) / 256, 256, 0, stream>>>(w, wh, wl, MMAX * LMAX * NLAT);
    sht_dft<<<dim3(4096, 2), 256, 0, stream>>>(x, cth, ctl, sth, stl, frh, frl, fih, fil);
    sht_leg<<<dim3(256, 32), 256, 0, stream>>>(frh, frl, fih, fil, wh, wl, out);
    (void)in_sizes; (void)n_in; (void)out_size; (void)ws_size;
}
